// GRU_encoder_decoder_7181185319035
// MI455X (gfx1250) — compile-verified, matrix operand order checked by reading, not run on hardware
//
#include <hip/hip_runtime.h>
#include <hip/hip_bf16.h>

typedef __attribute__((ext_vector_type(16))) _Float16 v16h;
typedef __attribute__((ext_vector_type(8)))  float    v8f;

// Problem constants (from reference)
constexpr int B_  = 1024;
constexpr int S_  = 64;
constexpr int T_  = 14;
constexpr int IN_ = 3;
constexpr int H_  = 1024;
constexpr int N3_ = 3 * H_;   // 3072

// GEMM tiling
constexpr int BM = 128;
constexpr int BN = 128;
constexpr int BK = 32;
constexpr int LDSPAD = 8;     // rows padded to 40 halfs = 80B (16B-aligned, conflict-free)

// ---------------------------------------------------------------------------
// C[M x N] (f32) = A[M x K] (f16, row-major) * Bw[N x K]^T (f16, row-major)
// Double-buffered LDS staging via CDNA5 async copies (ASYNCcnt-tracked):
//   global_load_async_to_lds_b128 writes memory->LDS with no VGPR round-trip.
// Steady-state loop: issue 4 async ops for chunk k+1, s_wait_asynccnt 4
// (chunk k complete, k+1 in flight), barrier, 8x WMMA, barrier.
// ---------------------------------------------------------------------------
__global__ __launch_bounds__(256) void gemm_wmma(
    const _Float16* __restrict__ A,
    const _Float16* __restrict__ Bw,
    float* __restrict__ C,
    int M, int N, int K)
{
    __shared__ _Float16 sA[2][BM][BK + LDSPAD];   // 2 x 10 KB
    __shared__ _Float16 sB[2][BN][BK + LDSPAD];   // 2 x 10 KB

    const int tid  = threadIdx.x;
    const int wid  = tid >> 5;
    const int lane = tid & 31;

    const int bm = blockIdx.y * BM;
    const int bn = blockIdx.x * BN;

    const int wm = (wid & 3) * 32;  // wave M offset (4 waves * 32 = 128)
    const int wn = (wid >> 2) * 64; // wave N offset (2 waves * 64 = 128)

    v8f acc[2][4];
#pragma unroll
    for (int i = 0; i < 2; ++i)
#pragma unroll
        for (int j = 0; j < 4; ++j)
            acc[i][j] = (v8f){0.f, 0.f, 0.f, 0.f, 0.f, 0.f, 0.f, 0.f};

    const int row = lane & 15;
    const int kh  = (lane >> 4) << 3;   // 0 or 8 (K-half of 16-bit WMMA operand layout)

    // One async b128 copy: global (src base, element index) -> LDS buffer.
    auto issue = [&](const _Float16* src, _Float16* ldsBase, int i, int k0, int K_) {
        const int r = i >> 2;
        const int c = (i & 3) * 8;
        const unsigned long long ga =
            (unsigned long long)(const void*)(src + (size_t)r * K_ + k0 + c);
        const unsigned la =
            (unsigned)(uintptr_t)(const void*)(ldsBase + r * (BK + LDSPAD) + c);
        asm volatile("global_load_async_to_lds_b128 %0, %1, off"
                     :: "v"(la), "v"(ga) : "memory");
    };

    // Stage one 128x32 A tile + 128x32 B tile: exactly 4 async ops per thread,
    // fully unrolled (no data-dependent control flow).
    auto stage = [&](int buf, int k0) {
        const _Float16* Abase = A  + (size_t)bm * K;
        const _Float16* Bbase = Bw + (size_t)bn * K;
        issue(Abase, &sA[buf][0][0], tid,       k0, K);
        issue(Abase, &sA[buf][0][0], tid + 256, k0, K);
        issue(Bbase, &sB[buf][0][0], tid,       k0, K);
        issue(Bbase, &sB[buf][0][0], tid + 256, k0, K);
    };

    // Consume buffer `buf`: 6 fragment loads (2xA, 4xB) + 8 WMMAs.
    auto compute = [&](int buf) {
        v16h af[2], bf[4];
#pragma unroll
        for (int mt = 0; mt < 2; ++mt) {
            const _Float16* p = &sA[buf][wm + mt * 16 + row][kh];
            ((uint4*)&af[mt])[0] = *(const uint4*)p;
            ((uint4*)&af[mt])[1] = *(const uint4*)(p + 16);
        }
#pragma unroll
        for (int nt = 0; nt < 4; ++nt) {
            const _Float16* p = &sB[buf][wn + nt * 16 + row][kh];
            ((uint4*)&bf[nt])[0] = *(const uint4*)p;
            ((uint4*)&bf[nt])[1] = *(const uint4*)(p + 16);
        }
#pragma unroll
        for (int mt = 0; mt < 2; ++mt)
#pragma unroll
            for (int nt = 0; nt < 4; ++nt)
                acc[mt][nt] = __builtin_amdgcn_wmma_f32_16x16x32_f16(
                    false, af[mt], false, bf[nt],
                    (short)0, acc[mt][nt], false, false);
    };

    const int nk = K / BK;
    stage(0, 0);                               // prime buffer 0

    // Steady state: nk-1 iterations, branch-free staging + wait 4.
    for (int kt = 0; kt < nk - 1; ++kt) {
        stage((kt + 1) & 1, (kt + 1) * BK);
        asm volatile("s_wait_asynccnt 0x4" ::: "memory");  // chunk kt landed
        __syncthreads();
        compute(kt & 1);
        __syncthreads();   // done reading; next stage may overwrite this buffer
    }

    // Epilogue: last chunk.
    asm volatile("s_wait_asynccnt 0x0" ::: "memory");
    __syncthreads();
    compute((nk - 1) & 1);

    // Store: C/D layout -> lane&15 = N, (lane>>4)*8 + r = M
#pragma unroll
    for (int mt = 0; mt < 2; ++mt) {
#pragma unroll
        for (int nt = 0; nt < 4; ++nt) {
            const int mbase = bm + wm + mt * 16 + (lane >> 4) * 8;
            const int nn    = bn + wn + nt * 16 + (lane & 15);
#pragma unroll
            for (int r = 0; r < 8; ++r)
                C[(size_t)(mbase + r) * N + nn] = acc[mt][nt][r];
        }
    }
}

// ---------------------------------------------------------------------------
// Elementwise GRU gate combine. Layer-0 variant folds the K=3 input GEMM.
// ---------------------------------------------------------------------------
__device__ __forceinline__ float sigmoidf_(float x) {
    return 1.f / (1.f + __expf(-x));
}

__global__ __launch_bounds__(256) void gru_combine_small(
    float* __restrict__ h, _Float16* __restrict__ h16,
    const float* __restrict__ Gh,
    const float* __restrict__ x, int x_stride, int x_off,
    const float* __restrict__ Wih,   // (3H x 3) f32
    const float* __restrict__ bih, const float* __restrict__ bhh)
{
    const int idx = blockIdx.x * 256 + threadIdx.x;   // over B*H
    const int b = idx >> 10;
    const int j = idx & (H_ - 1);

    const float x0 = x[(size_t)b * x_stride + x_off + 0];
    const float x1 = x[(size_t)b * x_stride + x_off + 1];
    const float x2 = x[(size_t)b * x_stride + x_off + 2];

    const float gir = Wih[(size_t)j * 3 + 0] * x0 + Wih[(size_t)j * 3 + 1] * x1 +
                      Wih[(size_t)j * 3 + 2] * x2 + bih[j];
    const float giz = Wih[(size_t)(H_ + j) * 3 + 0] * x0 + Wih[(size_t)(H_ + j) * 3 + 1] * x1 +
                      Wih[(size_t)(H_ + j) * 3 + 2] * x2 + bih[H_ + j];
    const float gin = Wih[(size_t)(2 * H_ + j) * 3 + 0] * x0 + Wih[(size_t)(2 * H_ + j) * 3 + 1] * x1 +
                      Wih[(size_t)(2 * H_ + j) * 3 + 2] * x2 + bih[2 * H_ + j];

    const size_t grow = (size_t)b * N3_;
    const float ghr = Gh[grow + j]           + bhh[j];
    const float ghz = Gh[grow + H_ + j]      + bhh[H_ + j];
    const float ghn = Gh[grow + 2 * H_ + j]  + bhh[2 * H_ + j];

    const float r = sigmoidf_(gir + ghr);
    const float z = sigmoidf_(giz + ghz);
    const float n = tanhf(gin + r * ghn);
    const float hp = h[idx];
    const float hnew = (1.f - z) * n + z * hp;
    h[idx]   = hnew;
    h16[idx] = (_Float16)hnew;
}

__global__ __launch_bounds__(256) void gru_combine_gi(
    float* __restrict__ h, _Float16* __restrict__ h16,
    const float* __restrict__ Gh, const float* __restrict__ Gi,
    const float* __restrict__ bih, const float* __restrict__ bhh)
{
    const int idx = blockIdx.x * 256 + threadIdx.x;
    const int b = idx >> 10;
    const int j = idx & (H_ - 1);
    const size_t grow = (size_t)b * N3_;

    const float gir = Gi[grow + j]          + bih[j];
    const float giz = Gi[grow + H_ + j]     + bih[H_ + j];
    const float gin = Gi[grow + 2 * H_ + j] + bih[2 * H_ + j];
    const float ghr = Gh[grow + j]          + bhh[j];
    const float ghz = Gh[grow + H_ + j]     + bhh[H_ + j];
    const float ghn = Gh[grow + 2 * H_ + j] + bhh[2 * H_ + j];

    const float r = sigmoidf_(gir + ghr);
    const float z = sigmoidf_(giz + ghz);
    const float n = tanhf(gin + r * ghn);
    const float hp = h[idx];
    const float hnew = (1.f - z) * n + z * hp;
    h[idx]   = hnew;
    h16[idx] = (_Float16)hnew;
}

// ---------------------------------------------------------------------------
// Decoder output head: out = h1 @ lin_W^T + b, plus autoregressive dinp update
// One wave (32 lanes) per batch row.
// ---------------------------------------------------------------------------
__global__ __launch_bounds__(256) void dec_out_kernel(
    const float* __restrict__ h1,
    const float* __restrict__ linW, const float* __restrict__ linb,
    float* __restrict__ dinp, float* __restrict__ out, int t)
{
    const int wid  = threadIdx.x >> 5;
    const int lane = threadIdx.x & 31;
    const int b    = blockIdx.x * 8 + wid;

    float s = 0.f;
    for (int k = lane; k < H_; k += 32)
        s += h1[(size_t)b * H_ + k] * linW[k];
#pragma unroll
    for (int off = 16; off > 0; off >>= 1)
        s += __shfl_xor(s, off, 32);

    if (lane == 0) {
        const float d0 = s + linb[0];
        out[(size_t)b * T_ + t] = d0;
        const float o0 = dinp[b * 3 + 0];
        const float o1 = dinp[b * 3 + 1];
        const float d1 = o0 - d0;
        const float d2 = o1 - d1;
        dinp[b * 3 + 0] = d0;
        dinp[b * 3 + 1] = d1;
        dinp[b * 3 + 2] = d2;
    }
}

// ---------------------------------------------------------------------------
// Utility kernels
// ---------------------------------------------------------------------------
__global__ __launch_bounds__(256) void conv_f32_f16(
    const float* __restrict__ w, _Float16* __restrict__ o, int n)
{
    const int i = blockIdx.x * 256 + threadIdx.x;
    if (i < n) o[i] = (_Float16)w[i];
}

__global__ __launch_bounds__(256) void zero_state(
    float* __restrict__ h, _Float16* __restrict__ h16, int n)
{
    const int i = blockIdx.x * 256 + threadIdx.x;
    if (i < n) { h[i] = 0.f; h16[i] = (_Float16)0.f; }
}

__global__ __launch_bounds__(256) void init_dinp_kernel(
    float* __restrict__ dinp, const float* __restrict__ x)
{
    const int i = blockIdx.x * 256 + threadIdx.x;   // B*3
    if (i < B_ * IN_) {
        const int b = i / 3;
        const int k = i % 3;
        dinp[i] = x[((size_t)b * S_ + (S_ - 1)) * IN_ + k];
    }
}

// ---------------------------------------------------------------------------
// Host driver
// ---------------------------------------------------------------------------
extern "C" void kernel_launch(void* const* d_in, const int* in_sizes, int n_in,
                              void* d_out, int out_size, void* d_ws, size_t ws_size,
                              hipStream_t stream)
{
    // setup_inputs() dict order
    const float* x        = (const float*)d_in[0];
    const float* enc_Wih0 = (const float*)d_in[4];
    const float* enc_Whh0 = (const float*)d_in[5];
    const float* enc_bih0 = (const float*)d_in[6];
    const float* enc_bhh0 = (const float*)d_in[7];
    const float* enc_Wih1 = (const float*)d_in[8];
    const float* enc_Whh1 = (const float*)d_in[9];
    const float* enc_bih1 = (const float*)d_in[10];
    const float* enc_bhh1 = (const float*)d_in[11];
    const float* dec_Wih0 = (const float*)d_in[12];
    const float* dec_Whh0 = (const float*)d_in[13];
    const float* dec_bih0 = (const float*)d_in[14];
    const float* dec_bhh0 = (const float*)d_in[15];
    const float* dec_Wih1 = (const float*)d_in[16];
    const float* dec_Whh1 = (const float*)d_in[17];
    const float* dec_bih1 = (const float*)d_in[18];
    const float* dec_bhh1 = (const float*)d_in[19];
    const float* lin_W    = (const float*)d_in[20];
    const float* lin_b    = (const float*)d_in[21];
    float* out = (float*)d_out;

    // Workspace carve-up (~76 MB)
    char* ws = (char*)d_ws;
    auto carve = [&](size_t bytes) -> void* {
        void* p = (void*)ws;
        ws += (bytes + 255) & ~(size_t)255;
        return p;
    };
    const size_t WELEMS = (size_t)N3_ * H_;   // 3072*1024
    _Float16* Wenc_hh0 = (_Float16*)carve(WELEMS * 2);
    _Float16* Wenc_ih1 = (_Float16*)carve(WELEMS * 2);
    _Float16* Wenc_hh1 = (_Float16*)carve(WELEMS * 2);
    _Float16* Wdec_hh0 = (_Float16*)carve(WELEMS * 2);
    _Float16* Wdec_ih1 = (_Float16*)carve(WELEMS * 2);
    _Float16* Wdec_hh1 = (_Float16*)carve(WELEMS * 2);
    float*    h0   = (float*)carve((size_t)B_ * H_ * 4);
    float*    h1   = (float*)carve((size_t)B_ * H_ * 4);
    _Float16* h0h  = (_Float16*)carve((size_t)B_ * H_ * 2);
    _Float16* h1h  = (_Float16*)carve((size_t)B_ * H_ * 2);
    float*    Gh   = (float*)carve((size_t)B_ * N3_ * 4);
    float*    Gi   = (float*)carve((size_t)B_ * N3_ * 4);
    float*    dinp = (float*)carve((size_t)B_ * IN_ * 4);

    // Weight conversion f32 -> f16 (native [n][k] layout, no transpose needed)
    const int wgrid = (int)((WELEMS + 255) / 256);
    conv_f32_f16<<<wgrid, 256, 0, stream>>>(enc_Whh0, Wenc_hh0, (int)WELEMS);
    conv_f32_f16<<<wgrid, 256, 0, stream>>>(enc_Wih1, Wenc_ih1, (int)WELEMS);
    conv_f32_f16<<<wgrid, 256, 0, stream>>>(enc_Whh1, Wenc_hh1, (int)WELEMS);
    conv_f32_f16<<<wgrid, 256, 0, stream>>>(dec_Whh0, Wdec_hh0, (int)WELEMS);
    conv_f32_f16<<<wgrid, 256, 0, stream>>>(dec_Wih1, Wdec_ih1, (int)WELEMS);
    conv_f32_f16<<<wgrid, 256, 0, stream>>>(dec_Whh1, Wdec_hh1, (int)WELEMS);

    const int hgrid = (B_ * H_) / 256;   // 4096
    zero_state<<<hgrid, 256, 0, stream>>>(h0, h0h, B_ * H_);
    zero_state<<<hgrid, 256, 0, stream>>>(h1, h1h, B_ * H_);
    init_dinp_kernel<<<(B_ * IN_ + 255) / 256, 256, 0, stream>>>(dinp, x);

    const dim3 ggrid(N3_ / BN, B_ / BM);   // (24, 8)

    // -------------------- Encoder --------------------
    for (int t = 0; t < S_; ++t) {
        // layer 0: Gh = h0 @ Whh0^T ; combine folds x@Wih0^T (K=3)
        gemm_wmma<<<ggrid, 256, 0, stream>>>(h0h, Wenc_hh0, Gh, B_, N3_, H_);
        gru_combine_small<<<hgrid, 256, 0, stream>>>(
            h0, h0h, Gh, x, S_ * IN_, t * IN_, enc_Wih0, enc_bih0, enc_bhh0);
        // layer 1: Gh = h1 @ Whh1^T (old h1) ; Gi = new h0 @ Wih1^T
        gemm_wmma<<<ggrid, 256, 0, stream>>>(h1h, Wenc_hh1, Gh, B_, N3_, H_);
        gemm_wmma<<<ggrid, 256, 0, stream>>>(h0h, Wenc_ih1, Gi, B_, N3_, H_);
        gru_combine_gi<<<hgrid, 256, 0, stream>>>(h1, h1h, Gh, Gi, enc_bih1, enc_bhh1);
    }

    // -------------------- Decoder --------------------
    for (int t = 0; t < T_; ++t) {
        gemm_wmma<<<ggrid, 256, 0, stream>>>(h0h, Wdec_hh0, Gh, B_, N3_, H_);
        gru_combine_small<<<hgrid, 256, 0, stream>>>(
            h0, h0h, Gh, dinp, IN_, 0, dec_Wih0, dec_bih0, dec_bhh0);
        gemm_wmma<<<ggrid, 256, 0, stream>>>(h1h, Wdec_hh1, Gh, B_, N3_, H_);
        gemm_wmma<<<ggrid, 256, 0, stream>>>(h0h, Wdec_ih1, Gi, B_, N3_, H_);
        gru_combine_gi<<<hgrid, 256, 0, stream>>>(h1, h1h, Gh, Gi, dec_bih1, dec_bhh1);
        dec_out_kernel<<<B_ / 8, 256, 0, stream>>>(h1, lin_W, lin_b, dinp, out, t);
    }
}
